// GATModel_77884936945980
// MI455X (gfx1250) — matrix operand order repaired and checked by reading, not compiled, not run on hardware
//
#include <hip/hip_runtime.h>
#include <math.h>

// ---------------------------------------------------------------------------
// GAT model for MI455X (gfx1250, wave32).
// Dense N x 64 x 64 GEMMs use v_wmma_f32_16x16x32_f16 (f16 in, f32 acc),
// with weight fragments held in registers across 4 row-tiles per block,
// float4 LDS staging and global_prefetch of the next tile.
// Edge passes are streaming kernels with float atomics (softmax scatter).
// Edge head is algebraically decomposed: concat(h[s],h[d],ea) @ W1 ==
//   (h@W1a)[s] + (h@W1b)[d] + ea @ W1c   -> per-node GEMMs + cheap edge pass.
// ---------------------------------------------------------------------------

typedef __attribute__((ext_vector_type(16))) _Float16 v16h;
typedef __attribute__((ext_vector_type(8)))  float    v8f;

#define HIDN 64
#define TPB_TILES 4   // row tiles (16 rows each) processed per block

__device__ __forceinline__ void atomic_max_f32(float* addr, float val) {
  // IEEE-ordered integer trick: works for any sign mix, init must be negative.
  if (val >= 0.0f) {
    atomicMax((int*)addr, __float_as_int(val));
  } else {
    atomicMin((unsigned int*)addr, (unsigned int)__float_as_int(val));
  }
}

// ---------------------------------------------------------------------------
// C[nrows,64] = act( A[nrows,64] @ W[64(ldw),64] + bias )
// Block = 128 threads = 4 waves; wave w owns cols [16w,16w+16).
// B fragments + bias are built once and reused across TPB_TILES row tiles.
// ---------------------------------------------------------------------------
__global__ void gemm64_wmma(const float* __restrict__ A,
                            const float* __restrict__ W, int ldw,
                            const float* __restrict__ bias, int act,
                            float* __restrict__ C, int nrows) {
  __shared__ __align__(16) float As[16 * HIDN];

  const int lane = threadIdx.x & 31;
  const int wave = threadIdx.x >> 5;        // 0..3: column tile
  const int m    = lane & 15;
  const int hi   = lane >> 4;               // 0 or 1 (half-wave)
  const int col  = wave * 16 + m;

  // B fragment: 32x16 f16, element e -> K = e + 16*hi, N = col. Built once.
  v16h b0, b1;
  const int bbase = hi * 16;
#pragma unroll
  for (int e = 0; e < 16; ++e) {
    int k = e + bbase;
    b0[e] = (_Float16)W[(size_t)k * ldw + col];
    b1[e] = (_Float16)W[(size_t)(k + 32) * ldw + col];
  }
  const float bv = bias ? bias[col] : 0.0f;

  const int ntiles = (nrows + 15) >> 4;
  const int tile0  = blockIdx.x * TPB_TILES;
  const float4* __restrict__ A4 = (const float4*)A;
  float4* As4 = (float4*)As;

  for (int t = 0; t < TPB_TILES; ++t) {
    const int tile = tile0 + t;
    if (tile >= ntiles) break;              // block-uniform
    const int row0 = tile * 16;

    __syncthreads();                        // As reuse fence
    // Stage 16x64 A tile: 256 float4s, 2 per thread (b128 loads).
    for (int i = threadIdx.x; i < 256; i += blockDim.x) {
      int r  = i >> 4;
      int gr = row0 + r;
      float4 v = make_float4(0.f, 0.f, 0.f, 0.f);
      if (gr < nrows) v = A4[(size_t)gr * 16 + (i & 15)];
      As4[i] = v;
    }
    __syncthreads();

    // Prefetch next row tile (global_prefetch_b8): 4KB = 32 x 128B lines.
    const int nrow0 = row0 + 16;
    if (nrow0 + 16 <= nrows && threadIdx.x < 32) {
      __builtin_prefetch((const char*)(A + (size_t)nrow0 * HIDN) +
                             threadIdx.x * 128, 0, 0);
    }

    // A fragment: 16x32 f16, lane m holds row M=m; element e -> K = e + 16*hi.
    v16h a0, a1;
    const int abase = hi * 16;
#pragma unroll
    for (int e = 0; e < 16; ++e) {
      int k = e + abase;       // K in 0..31
      a0[e] = (_Float16)As[m * HIDN + k];
      a1[e] = (_Float16)As[m * HIDN + 32 + k];
    }

    v8f acc = {};
    acc = __builtin_amdgcn_wmma_f32_16x16x32_f16(false, a0, false, b0,
                                                 (short)0, acc, false, false);
    acc = __builtin_amdgcn_wmma_f32_16x16x32_f16(false, a1, false, b1,
                                                 (short)0, acc, false, false);

    if (row0 + 16 <= nrows) {               // full tile: unguarded epilogue
#pragma unroll
      for (int r = 0; r < 8; ++r) {
        float v = acc[r] + bv;
        if (act == 1) v = (v > 0.0f) ? v : 0.01f * v;
        C[(size_t)(row0 + r + hi * 8) * HIDN + col] = v;
      }
    } else {                                // tail tile
#pragma unroll
      for (int r = 0; r < 8; ++r) {
        int row = row0 + r + hi * 8;
        if (row < nrows) {
          float v = acc[r] + bv;
          if (act == 1) v = (v > 0.0f) ? v : 0.01f * v;
          C[(size_t)row * HIDN + col] = v;
        }
      }
    }
  }
}

// h0[n,c] = x[n,0]*W[0,c] + x[n,1]*W[1,c] + b[c]
__global__ void encoder_kernel(const float* __restrict__ x,
                               const float* __restrict__ W,
                               const float* __restrict__ b,
                               float* __restrict__ h0, int N) {
  int tid = blockIdx.x * blockDim.x + threadIdx.x;
  if (tid >= N * HIDN) return;
  int n = tid >> 6, c = tid & 63;
  h0[tid] = x[(size_t)n * 2] * W[c] + x[(size_t)n * 2 + 1] * W[HIDN + c] + b[c];
}

// alpha_src[n,h] = sum_c h1[n,h*C+c]*a_s[h,c];  same for alpha_dst
__global__ void alpha_kernel(const float* __restrict__ h1,
                             const float* __restrict__ a_s,
                             const float* __restrict__ a_d,
                             int heads, int C,
                             float* __restrict__ asrc,
                             float* __restrict__ adst, int N) {
  int n = blockIdx.x * blockDim.x + threadIdx.x;
  if (n >= N) return;
  for (int h = 0; h < heads; ++h) {
    float sa = 0.0f, sd = 0.0f;
    for (int c = 0; c < C; ++c) {
      float v = h1[(size_t)n * HIDN + h * C + c];
      sa += v * a_s[h * C + c];
      sd += v * a_d[h * C + c];
    }
    asrc[(size_t)n * heads + h] = sa;
    adst[(size_t)n * heads + h] = sd;
  }
}

__global__ void gat_init_kernel(float* __restrict__ amax,
                                float* __restrict__ asum,
                                float* __restrict__ acc, int N, int heads) {
  int tid = blockIdx.x * blockDim.x + threadIdx.x;
  if (tid < N * HIDN) acc[tid] = 0.0f;
  if (tid < N * heads) { amax[tid] = -3.0e38f; asum[tid] = 0.0f; }
}

// Pass 1: amax[d,h] = max over incoming edges (+ self loop) of lrelu_0.2
__global__ void edge_max_kernel(const int* __restrict__ ei,
                                const float* __restrict__ asrc,
                                const float* __restrict__ adst,
                                float* __restrict__ amax,
                                int E, int N, int heads) {
  int tid = blockIdx.x * blockDim.x + threadIdx.x;
  int total = (E + N) * heads;
  if (tid >= total) return;
  int e = tid / heads, h = tid - e * heads;
  int s, d;
  if (e < E) { s = ei[e]; d = ei[E + e]; } else { s = d = e - E; }
  float v = asrc[(size_t)s * heads + h] + adst[(size_t)d * heads + h];
  v = (v > 0.0f) ? v : 0.2f * v;
  atomic_max_f32(&amax[(size_t)d * heads + h], v);
}

// Pass 2: asum[d,h] += exp(a - amax);  acc[d,t] += exp * h1[s,t]
__global__ void edge_acc_kernel(const int* __restrict__ ei,
                                const float* __restrict__ h1,
                                const float* __restrict__ asrc,
                                const float* __restrict__ adst,
                                const float* __restrict__ amax,
                                float* __restrict__ asum,
                                float* __restrict__ acc,
                                int E, int N, int heads, int C) {
  long long tid = (long long)blockIdx.x * blockDim.x + threadIdx.x;
  long long total = (long long)(E + N) * HIDN;
  if (tid >= total) return;
  int e = (int)(tid >> 6), t = (int)(tid & 63);
  int h = t / C;
  int s, d;
  if (e < E) { s = ei[e]; d = ei[E + e]; } else { s = d = e - E; }
  float a = asrc[(size_t)s * heads + h] + adst[(size_t)d * heads + h];
  a = (a > 0.0f) ? a : 0.2f * a;
  float ex = __expf(a - amax[(size_t)d * heads + h]);
  atomicAdd(&acc[(size_t)d * HIDN + t], ex * h1[(size_t)s * HIDN + t]);
  if ((t & (C - 1)) == 0) atomicAdd(&asum[(size_t)d * heads + h], ex);
}

// Pass 3: out[n,t] = acc[n,t]/(asum[n,h]+1e-16) + b[t]  (optional ELU)
__global__ void gat_norm_kernel(const float* __restrict__ acc,
                                const float* __restrict__ asum,
                                const float* __restrict__ b,
                                float* __restrict__ dst,
                                int N, int heads, int C, int elu) {
  int tid = blockIdx.x * blockDim.x + threadIdx.x;
  if (tid >= N * HIDN) return;
  int n = tid >> 6, t = tid & 63;
  int h = t / C;
  float v = acc[tid] / (asum[(size_t)n * heads + h] + 1e-16f) + b[t];
  if (elu) v = (v > 0.0f) ? v : expm1f(v);
  dst[tid] = v;
}

// node_pred[n,:] = tanh(t[n,:] @ W2[64,2] + b2)
__global__ void node_out_kernel(const float* __restrict__ t,
                                const float* __restrict__ W2,
                                const float* __restrict__ b2,
                                float* __restrict__ out, int N) {
  int n = blockIdx.x * blockDim.x + threadIdx.x;
  if (n >= N) return;
  float a0 = b2[0], a1 = b2[1];
  for (int c = 0; c < HIDN; ++c) {
    float v = t[(size_t)n * HIDN + c];
    a0 += v * W2[c * 2 + 0];
    a1 += v * W2[c * 2 + 1];
  }
  out[(size_t)n * 2 + 0] = tanhf(a0);
  out[(size_t)n * 2 + 1] = tanhf(a1);
}

// edge_pred[e] = tanh( lrelu(u[s]+v[d]+ea@W1c+b1, 0.01) . W2 + b2 )
// One wave per edge; lane handles channels c and c+32; wave32 reduction.
__global__ void edge_out_kernel(const int* __restrict__ ei,
                                const float* __restrict__ eattr,
                                const float* __restrict__ U,
                                const float* __restrict__ V,
                                const float* __restrict__ W1c,  // [2,64]
                                const float* __restrict__ b1,   // [64]
                                const float* __restrict__ W2,   // [64]
                                const float* __restrict__ b2,   // [1]
                                float* __restrict__ out, int E) {
  int gw   = (int)(((long long)blockIdx.x * blockDim.x + threadIdx.x) >> 5);
  int lane = threadIdx.x & 31;
  if (gw >= E) return;                       // uniform per wave
  int s = ei[gw];
  int d = ei[E + gw];
  float ea0 = eattr[(size_t)gw * 2 + 0];
  float ea1 = eattr[(size_t)gw * 2 + 1];
  float sum = 0.0f;
#pragma unroll
  for (int r = 0; r < 2; ++r) {
    int c = lane + r * 32;
    float z = U[(size_t)s * HIDN + c] + V[(size_t)d * HIDN + c]
            + ea0 * W1c[c] + ea1 * W1c[HIDN + c] + b1[c];
    z = (z > 0.0f) ? z : 0.01f * z;
    sum += z * W2[c];
  }
#pragma unroll
  for (int msk = 16; msk >= 1; msk >>= 1) sum += __shfl_xor(sum, msk, 32);
  if (lane == 0) out[gw] = tanhf(sum + b2[0]);
}

// ---------------------------------------------------------------------------
extern "C" void kernel_launch(void* const* d_in, const int* in_sizes, int n_in,
                              void* d_out, int out_size, void* d_ws, size_t ws_size,
                              hipStream_t stream) {
  const float* x      = (const float*)d_in[0];
  const int*   ei     = (const int*)  d_in[1];
  const float* eattr  = (const float*)d_in[2];
  const float* enc_W  = (const float*)d_in[3];
  const float* enc_b  = (const float*)d_in[4];
  const float* g1_W   = (const float*)d_in[5];
  const float* g1_as  = (const float*)d_in[6];
  const float* g1_ad  = (const float*)d_in[7];
  const float* g1_b   = (const float*)d_in[8];
  const float* g2_W   = (const float*)d_in[9];
  const float* g2_as  = (const float*)d_in[10];
  const float* g2_ad  = (const float*)d_in[11];
  const float* g2_b   = (const float*)d_in[12];
  const float* no_W1  = (const float*)d_in[13];
  const float* no_b1  = (const float*)d_in[14];
  const float* no_W2  = (const float*)d_in[15];
  const float* no_b2  = (const float*)d_in[16];
  const float* eo_W1  = (const float*)d_in[17];  // [130,64]
  const float* eo_b1  = (const float*)d_in[18];
  const float* eo_W2  = (const float*)d_in[19];  // [64,1]
  const float* eo_b2  = (const float*)d_in[20];

  const int N = in_sizes[0] / 2;      // x is [N,2]
  const int E = in_sizes[1] / 2;      // edge_index is [2,E]

  float* buf0 = (float*)d_ws;                         // [N,64]
  float* buf1 = buf0 + (size_t)N * HIDN;              // [N,64]
  float* buf2 = buf1 + (size_t)N * HIDN;              // [N,64]
  float* asrc = buf2 + (size_t)N * HIDN;              // [N,4]
  float* adst = asrc + (size_t)N * 4;                 // [N,4]
  float* amax = adst + (size_t)N * 4;                 // [N,4]
  float* asum = amax + (size_t)N * 4;                 // [N,4]

  float* edge_pred = (float*)d_out;                   // [E]
  float* node_pred = edge_pred + E;                   // [N,2]

  const int B = 256;
  const int ntiles = (N + 15) / 16;
  const int gblk   = (ntiles + TPB_TILES - 1) / TPB_TILES;
  const int nNH    = (N * HIDN + B - 1) / B;
  const int nN     = (N + B - 1) / B;

  // ---- node encoder: buf0 = x @ enc_W + enc_b
  encoder_kernel<<<nNH, B, 0, stream>>>(x, enc_W, enc_b, buf0, N);

  // ---- GAT layer 1 (heads=4, C=16), ELU
  gemm64_wmma<<<gblk, 128, 0, stream>>>(buf0, g1_W, HIDN, nullptr, 0, buf1, N);
  alpha_kernel<<<nN, B, 0, stream>>>(buf1, g1_as, g1_ad, 4, 16, asrc, adst, N);
  gat_init_kernel<<<nNH, B, 0, stream>>>(amax, asum, buf2, N, 4);
  {
    long long tm = (long long)(E + N) * 4;
    edge_max_kernel<<<(int)((tm + B - 1) / B), B, 0, stream>>>(
        ei, asrc, adst, amax, E, N, 4);
    long long ta = (long long)(E + N) * HIDN;
    edge_acc_kernel<<<(int)((ta + B - 1) / B), B, 0, stream>>>(
        ei, buf1, asrc, adst, amax, asum, buf2, E, N, 4, 16);
  }
  gat_norm_kernel<<<nNH, B, 0, stream>>>(buf2, asum, g1_b, buf2, N, 4, 16, 1);

  // ---- GAT layer 2 (heads=1, C=64)
  gemm64_wmma<<<gblk, 128, 0, stream>>>(buf2, g2_W, HIDN, nullptr, 0, buf1, N);
  alpha_kernel<<<nN, B, 0, stream>>>(buf1, g2_as, g2_ad, 1, HIDN, asrc, adst, N);
  gat_init_kernel<<<nNH, B, 0, stream>>>(amax, asum, buf0, N, 1);
  {
    long long tm = (long long)(E + N);
    edge_max_kernel<<<(int)((tm + B - 1) / B), B, 0, stream>>>(
        ei, asrc, adst, amax, E, N, 1);
    long long ta = (long long)(E + N) * HIDN;
    edge_acc_kernel<<<(int)((ta + B - 1) / B), B, 0, stream>>>(
        ei, buf1, asrc, adst, amax, asum, buf0, E, N, 1, HIDN);
  }
  gat_norm_kernel<<<nNH, B, 0, stream>>>(buf0, asum, g2_b, buf0, N, 1, HIDN, 0);
  // buf0 = final node embedding h

  // ---- node head: node_pred = tanh(lrelu(h@W1+b1) @ W2 + b2)
  gemm64_wmma<<<gblk, 128, 0, stream>>>(buf0, no_W1, HIDN, no_b1, 1, buf1, N);
  node_out_kernel<<<nN, B, 0, stream>>>(buf1, no_W2, no_b2, node_pred, N);

  // ---- edge head (decomposed): u = h@W1a -> buf1, v = h@W1b -> buf2
  gemm64_wmma<<<gblk, 128, 0, stream>>>(buf0, eo_W1, HIDN, nullptr, 0, buf1, N);
  gemm64_wmma<<<gblk, 128, 0, stream>>>(buf0, eo_W1 + 64 * HIDN, HIDN,
                                        nullptr, 0, buf2, N);
  {
    long long te = (long long)E * 32;
    edge_out_kernel<<<(int)((te + B - 1) / B), B, 0, stream>>>(
        ei, eattr, buf1, buf2, eo_W1 + 128 * HIDN, eo_b1, eo_W2, eo_b2,
        edge_pred, E);
  }
}
